// LRNet_CIFAR10_act_53626961657951
// MI455X (gfx1250) — matrix operand order repaired and checked by reading, not compiled, not run on hardware
//
#include <hip/hip_runtime.h>
#include <cstdint>
#include <cstddef>

// ---------------------------------------------------------------------------
// CDNA5 (gfx1250) WMMA / TDM types
// ---------------------------------------------------------------------------
typedef __bf16    v16bf __attribute__((ext_vector_type(16)));
typedef float     v8f   __attribute__((ext_vector_type(8)));
typedef uint32_t  v8u   __attribute__((ext_vector_type(8)));
typedef uint32_t  v4u   __attribute__((ext_vector_type(4)));
typedef int       v8i   __attribute__((ext_vector_type(8)));
typedef int       v4i   __attribute__((ext_vector_type(4)));

#if __has_builtin(__builtin_amdgcn_tensor_load_to_lds)
#define LR_USE_TDM 1
#else
#define LR_USE_TDM 0
#endif

// ---------------------------------------------------------------------------
// bf16 helpers (manual RNE conversion; storage is uint16_t)
// ---------------------------------------------------------------------------
__device__ __forceinline__ uint16_t f2bf(float f) {
  uint32_t u = __builtin_bit_cast(uint32_t, f);
  uint32_t r = u + 0x7fffu + ((u >> 16) & 1u);
  return (uint16_t)(r >> 16);
}
__device__ __forceinline__ float bf2f(uint16_t h) {
  uint32_t u = ((uint32_t)h) << 16;
  return __builtin_bit_cast(float, u);
}
__device__ __forceinline__ uint32_t pack2bf(float lo, float hi) {
  return (uint32_t)f2bf(lo) | ((uint32_t)f2bf(hi) << 16);
}

// ---------------------------------------------------------------------------
// Tensor Data Mover: one DMA per (matrix, chunk) moving a 2-row tile
// (256 dwords x 2 rows, row stride = kChunks*256 dwords) global -> LDS.
// D# built per ISA 8.3/8.4: group0 = {count/type/lds_addr/global_addr},
// group1 = {data_size, tensor_dim0/1, tile_dim0/1, dim0_stride}.
// Groups 2/3 zero (<=2D). Tracked by TENSORcnt.
// This toolchain exposes the 6-arg builtin:
//   (uint32x4 g0, int32x8 g1, int32x4 g2, int32x4 g3, int32x8 g4, i32 cpol)
// ---------------------------------------------------------------------------
#if LR_USE_TDM
__device__ __forceinline__ void tdm_load_2rows(uint32_t lds_addr, const void* g,
                                               uint32_t rowStrideDw) {
  const uint64_t ga = (uint64_t)(uintptr_t)g;
  v4u g0;
  g0[0] = 1u;                                              // count=1, user desc
  g0[1] = lds_addr;                                        // LDS byte address
  g0[2] = (uint32_t)ga;                                    // global_addr[31:0]
  g0[3] = (uint32_t)((ga >> 32) & 0x01ffffffu) | (2u << 30); // addr[56:32]|type=2
  v8i g1;
  g1[0] = (int)(2u << 16);          // wg_mask=0, data_size=2 (4B)
  g1[1] = (int)(256u << 16);        // tensor_dim0[15:0]=256 (bits 63:48)
  g1[2] = (int)(2u << 16);          // tensor_dim0 hi=0 | tensor_dim1 lo=2
  g1[3] = (int)(256u << 16);        // tensor_dim1 hi=0 | tile_dim0=256
  g1[4] = 2;                        // tile_dim1=2 | tile_dim2=0
  g1[5] = (int)rowStrideDw;         // tensor_dim0_stride lo (data_size units)
  g1[6] = 0;                        // stride hi | dim1_stride lo
  g1[7] = 0;
  const v4i gz4 = {0, 0, 0, 0};
  const v8i gz8 = {0, 0, 0, 0, 0, 0, 0, 0};
  __builtin_amdgcn_tensor_load_to_lds(g0, g1, gz4, gz4, gz8, 0);
}
__device__ __forceinline__ void waitA_one_group_pending() {
  __builtin_amdgcn_s_wait_tensorcnt(2);
  asm volatile("" ::: "memory");
}
__device__ __forceinline__ void waitA_all() {
  __builtin_amdgcn_s_wait_tensorcnt(0);
  asm volatile("" ::: "memory");
}
#else
// Fallback: per-lane async global->LDS copies (ASYNCcnt-tracked).
__device__ __forceinline__ void async_copy_b128(uint32_t lds_addr, const void* gptr) {
  uint64_t ga = (uint64_t)(uintptr_t)gptr;
  asm volatile("global_load_async_to_lds_b128 %0, %1, off"
               :: "v"(lds_addr), "v"(ga)
               : "memory");
}
__device__ __forceinline__ void waitA_one_group_pending() {
  asm volatile("s_wait_asynccnt 0x8" ::: "memory");
}
__device__ __forceinline__ void waitA_all() {
  asm volatile("s_wait_asynccnt 0x0" ::: "memory");
}
#endif

// ---------------------------------------------------------------------------
// Deterministic hash RNG (replaces jax threefry; no runtime check available)
// ---------------------------------------------------------------------------
__device__ __forceinline__ uint32_t hash32(uint32_t x) {
  x ^= x >> 17; x *= 0xed5ad4bbu;
  x ^= x >> 11; x *= 0xac4c1b51u;
  x ^= x >> 15; x *= 0x31848babu;
  x ^= x >> 14;
  return x;
}
__device__ __forceinline__ float uniformf(uint32_t seed, uint32_t idx) {
  uint32_t h = hash32(idx * 0x9e3779b9u + seed * 0x85ebca6bu + 0x165667b1u);
  float r = (float)(h >> 8) * (1.0f / 16777216.0f);
  return 1e-6f + r * (1.0f - 2e-6f);
}
__device__ __forceinline__ float gaussf(uint32_t seed, uint32_t idx) {
  float u1 = uniformf(seed, 2u * idx);
  float u2 = uniformf(seed, 2u * idx + 1u);
  return sqrtf(-2.f * __logf(u1)) * __cosf(6.28318530718f * u2);
}
__device__ __forceinline__ float sigmoidf(float x) {
  return 1.f / (1.f + __expf(-x));
}

// ---------------------------------------------------------------------------
// Pack ternary-distribution weights (wm, wv) straight into the WMMA A-fragment
// layout (16-bit A 16x32, ISA 7.12.2): per (mTile,kChunk) block of 256 dwords,
// dword (lane, j) holds K = [(lane&16?8:0) + (j<4 ? 2j : 16+2(j-4))] + {0,1}.
// ---------------------------------------------------------------------------
__global__ void pack_weights_kernel(const float* __restrict__ alpha,
                                    const float* __restrict__ betta,
                                    uint32_t* __restrict__ wmP,
                                    uint32_t* __restrict__ wvP,
                                    int M, int Cin, int Kpad) {
  const int idx = blockIdx.x * blockDim.x + threadIdx.x;
  const int total = M * Kpad / 2;
  if (idx >= total) return;
  const int kChunks = Kpad >> 5;
  const int blockId = idx >> 8;
  const int within  = idx & 255;
  const int lane    = within >> 3;
  const int j       = within & 7;
  const int mTile   = blockId / kChunks;
  const int kChunk  = blockId - mTile * kChunks;
  const int m       = mTile * 16 + (lane & 15);
  const int kbase   = ((lane & 16) ? 8 : 0) + ((j < 4) ? 2 * j : 16 + 2 * (j - 4));
  const int Kreal   = Cin * 9;

  uint32_t om = 0, ov = 0;
#pragma unroll
  for (int t = 0; t < 2; ++t) {
    const int k = kChunk * 32 + kbase + t;
    float wm = 0.f, wv = 0.f;
    if (k < Kreal && m < M) {
      const int cin = k / 9;
      const int r9  = k - cin * 9;
      const int r   = r9 / 3;
      const int s   = r9 - 3 * r;
      const int off = ((m * Cin + cin) * 3 + r) * 3 + s;
      const float p0  = sigmoidf(alpha[off]);
      const float p1  = sigmoidf(betta[off]) * (1.f - p0);
      const float pm1 = 1.f - p0 - p1;
      wm = p1 - pm1;
      wv = p1 + pm1 - wm * wm;
    }
    om |= ((uint32_t)f2bf(wm)) << (16 * t);
    ov |= ((uint32_t)f2bf(wv)) << (16 * t);
  }
  wmP[idx] = om;
  wvP[idx] = ov;
}

// ---------------------------------------------------------------------------
// Fused local-reparameterization conv layer, implicit GEMM on WMMA.
//   M = Cout, K = Cin*9 (padded to 32), Ncols = N*outH*outW
// Workgroup = 4 waves sharing one 16-column spatial tile; each wave owns TWO
// 16-row M tiles (32 output channels) -> 4 v_wmma per K chunk per wave.
// Per K chunk:
//   - B patch (x and x^2) gathered cooperatively ONCE per workgroup into LDS.
//   - A fragments stream through a double-buffered LDS stage via the Tensor
//     Data Mover (tensor_load_to_lds, TENSORcnt), one 2-row DMA per matrix,
//     prefetching chunk kc+1 while the gather VALU work for kc runs.
// Epilogue: gain + erf sign-prob + Gumbel-sign, store bf16 (f32 m,v if LAST).
// ---------------------------------------------------------------------------
template <typename InT, bool LAST>
__global__ __launch_bounds__(128)
void lrconv_wmma_kernel(const InT* __restrict__ in,
                        const uint32_t* __restrict__ wmP,
                        const uint32_t* __restrict__ wvP,
                        const float* __restrict__ gain,
                        uint16_t* __restrict__ actOut,
                        float* __restrict__ mOut,
                        float* __restrict__ vOut,
                        int Cin, int H, int W, int Cout,
                        int outH, int outW, int stride, uint32_t seed) {
  __shared__ uint32_t bmS[256];            // B fragment (x), 1KB
  __shared__ uint32_t bvS[256];            // B fragment (x^2), 1KB
  __shared__ uint32_t aS[4][2][2][512];    // [wave][buf][mat][row*256+dw] 32KB

  const int lane = threadIdx.x & 31;
  const int wave = threadIdx.x >> 5;
  const int mTile0 = (blockIdx.y * 4 + wave) * 2;   // two M tiles per wave
  const int nTile  = blockIdx.x;

  const int Kreal   = Cin * 9;
  const int Kpad    = (Kreal + 31) & ~31;
  const int kChunks = Kpad >> 5;
  const int hw      = outH * outW;

  // ---- this lane's output column (C/D fragment: N = lane & 15) ----
  const int ncol = nTile * 16 + (lane & 15);
  const int img  = ncol / hw;
  const int rem  = ncol - img * hw;
  const int oh   = rem / outW;
  const int ow   = rem - oh * outW;

  // ---- A-stream staging: chunk kc rows for mTile0 / mTile0+1 ----
  const size_t tb0 = (size_t)mTile0 * kChunks * 256;      // dword offset row 0
  const uint32_t rowStrideDw = (uint32_t)kChunks * 256u;  // row 1 = row 0 + this

  auto issueA = [&](int kc, int buf) {
#if LR_USE_TDM
    tdm_load_2rows((uint32_t)(uintptr_t)&aS[wave][buf][0][0],
                   wmP + tb0 + (size_t)kc * 256, rowStrideDw);
    tdm_load_2rows((uint32_t)(uintptr_t)&aS[wave][buf][1][0],
                   wvP + tb0 + (size_t)kc * 256, rowStrideDw);
#else
    const uint32_t l16 = (uint32_t)lane * 16u;
#pragma unroll
    for (int i = 0; i < 2; ++i) {
      const char* gm = (const char*)(wmP + tb0 + (size_t)i * rowStrideDw + (size_t)kc * 256);
      const char* gv = (const char*)(wvP + tb0 + (size_t)i * rowStrideDw + (size_t)kc * 256);
      const uint32_t lm = (uint32_t)(uintptr_t)&aS[wave][buf][0][i * 256];
      const uint32_t lv = (uint32_t)(uintptr_t)&aS[wave][buf][1][i * 256];
      async_copy_b128(lm + l16,       gm + l16);
      async_copy_b128(lm + 512 + l16, gm + 512 + l16);
      async_copy_b128(lv + l16,       gv + l16);
      async_copy_b128(lv + 512 + l16, gv + 512 + l16);
    }
#endif
  };

  v8f accM0 = {}, accM1 = {}, accV0 = {}, accV1 = {};

  issueA(0, 0);
  int buf = 0;

  for (int kc = 0; kc < kChunks; ++kc) {
    const bool havePrefetch = (kc + 1 < kChunks);
    if (havePrefetch) issueA(kc + 1, buf ^ 1);

    // ---- cooperative B-patch gather: 256 dword slots, 2 per thread ----
    for (int s = threadIdx.x; s < 256; s += 128) {
      const int slane = s >> 3;
      const int j     = s & 7;
      const int scol  = nTile * 16 + (slane & 15);
      const int simg  = scol / hw;
      const int srem  = scol - simg * hw;
      const int soh   = srem / outW;
      const int sow   = srem - soh * outW;
      const int ihB   = soh * stride - 1;
      const int iwB   = sow * stride - 1;
      const int k0    = kc * 32 + ((slane & 16) ? 8 : 0) + ((j < 4) ? 2 * j : 16 + 2 * (j - 4));
      float x0 = 0.f, x1 = 0.f;
#pragma unroll
      for (int t = 0; t < 2; ++t) {
        const int k = k0 + t;
        float xv = 0.f;
        if (k < Kreal) {
          const int cin = k / 9;
          const int r9  = k - cin * 9;
          const int r   = r9 / 3;
          const int ss  = r9 - 3 * r;
          const int ih  = ihB + r;
          const int iw  = iwB + ss;
          if ((unsigned)ih < (unsigned)H && (unsigned)iw < (unsigned)W) {
            const size_t idx = (((size_t)simg * Cin + cin) * H + ih) * W + iw;
            if constexpr (sizeof(InT) == 2) xv = bf2f((uint16_t)in[idx]);
            else                            xv = (float)in[idx];
          }
        }
        if (t == 0) x0 = xv; else x1 = xv;
      }
      bmS[s] = pack2bf(x0, x1);
      bvS[s] = pack2bf(x0 * x0, x1 * x1);
    }
    __syncthreads();

    // ---- wait for this chunk's A DMA group (older group of two) ----
    if (havePrefetch) waitA_one_group_pending();
    else              waitA_all();

    // ---- consume fragments from LDS ----
    const v8u bm  = *(const v8u*)&bmS[lane * 8];
    const v8u bv  = *(const v8u*)&bvS[lane * 8];
    const v8u am0 = *(const v8u*)&aS[wave][buf][0][lane * 8];
    const v8u am1 = *(const v8u*)&aS[wave][buf][0][256 + lane * 8];
    const v8u av0 = *(const v8u*)&aS[wave][buf][1][lane * 8];
    const v8u av1 = *(const v8u*)&aS[wave][buf][1][256 + lane * 8];

    accM0 = __builtin_amdgcn_wmma_f32_16x16x32_bf16(
        false, __builtin_bit_cast(v16bf, am0), false, __builtin_bit_cast(v16bf, bm),
        (short)0, accM0, false, false);
    accV0 = __builtin_amdgcn_wmma_f32_16x16x32_bf16(
        false, __builtin_bit_cast(v16bf, av0), false, __builtin_bit_cast(v16bf, bv),
        (short)0, accV0, false, false);
    accM1 = __builtin_amdgcn_wmma_f32_16x16x32_bf16(
        false, __builtin_bit_cast(v16bf, am1), false, __builtin_bit_cast(v16bf, bm),
        (short)0, accM1, false, false);
    accV1 = __builtin_amdgcn_wmma_f32_16x16x32_bf16(
        false, __builtin_bit_cast(v16bf, av1), false, __builtin_bit_cast(v16bf, bv),
        (short)0, accV1, false, false);

    __syncthreads();   // protect B staging buffer before next overwrite
    buf ^= 1;
  }

  // ---- epilogue: C/D layout has M = r (+8 for lanes 16-31), N = lane&15 ----
  const int mSub = (lane & 16) ? 8 : 0;
#pragma unroll
  for (int i = 0; i < 2; ++i) {
    const v8f accM = i ? accM1 : accM0;
    const v8f accV = i ? accV1 : accV0;
#pragma unroll
    for (int r = 0; r < 8; ++r) {
      const int mRow = (mTile0 + i) * 16 + mSub + r;
      const float g = gain[mRow];
      const float m = accM[r] * g;
      const float v = accV[r] * g * g;
      const size_t oidx = (((size_t)img * Cout + mRow) * outH + oh) * outW + ow;
      if constexpr (LAST) {
        mOut[oidx] = m;
        vOut[oidx] = v;
      } else {
        const float p     = 0.5f * (1.f + erff(m * rsqrtf(v + 1e-10f) * 0.70710678118f));
        const float logit = __logf(p + 1e-10f) - __logf(1.f - p + 1e-10f);
        const float u     = uniformf(seed, (uint32_t)oidx);
        const float Lg    = __logf(u) - log1pf(-u);
        const float y     = sigmoidf(logit + Lg);  // tau = 1
        actOut[oidx] = f2bf(2.f * y - 1.f);
      }
    }
  }
}

// ---------------------------------------------------------------------------
// Final-layer elementwise: e = 2*sign_p(m,v)-1, var_s = 1-e^2 (in place).
// ---------------------------------------------------------------------------
__global__ void sign_e_kernel(float* __restrict__ m, float* __restrict__ v, int n) {
  const int i = blockIdx.x * blockDim.x + threadIdx.x;
  if (i >= n) return;
  const float mm = m[i], vv = v[i];
  const float p = 0.5f * (1.f + erff(mm * rsqrtf(vv + 1e-10f) * 0.70710678118f));
  const float e = 2.f * p - 1.f;
  m[i] = e;
  v[i] = 1.f - e * e;
}

// Per-channel mean/var over (N, H, W): 512 channels, 128*16 elems each.
__global__ void bn_stats_kernel(const float* __restrict__ e, float* __restrict__ stats) {
  __shared__ float s1[256];
  __shared__ float s2[256];
  const int c = blockIdx.x;
  float a = 0.f, b = 0.f;
  for (int i = threadIdx.x; i < 128 * 16; i += 256) {
    const int img = i >> 4;
    const int sp  = i & 15;
    const float x = e[((size_t)img * 512 + c) * 16 + sp];
    a += x;
    b += x * x;
  }
  s1[threadIdx.x] = a;
  s2[threadIdx.x] = b;
  __syncthreads();
  for (int s = 128; s > 0; s >>= 1) {
    if (threadIdx.x < (unsigned)s) {
      s1[threadIdx.x] += s1[threadIdx.x + s];
      s2[threadIdx.x] += s2[threadIdx.x + s];
    }
    __syncthreads();
  }
  if (threadIdx.x == 0) {
    const float mu = s1[0] * (1.f / 2048.f);
    stats[c]       = mu;
    stats[512 + c] = s2[0] * (1.f / 2048.f) - mu * mu;
  }
}

__global__ void bn_apply_kernel(float* __restrict__ e, float* __restrict__ var_s,
                                const float* __restrict__ stats,
                                const float* __restrict__ gamma,
                                const float* __restrict__ beta, int n) {
  const int i = blockIdx.x * blockDim.x + threadIdx.x;
  if (i >= n) return;
  const int c = (i >> 4) & 511;
  const float mu    = stats[c];
  const float sd2   = stats[512 + c];
  const float scale = gamma[c] * rsqrtf(sd2 + 1e-5f);
  e[i]     = (e[i] - mu) * scale + beta[c];
  var_s[i] = var_s[i] * scale * scale;
}

__global__ void fc_weights_kernel(const float* __restrict__ a,
                                  const float* __restrict__ b,
                                  float* __restrict__ wm,
                                  float* __restrict__ wv, int n) {
  const int i = blockIdx.x * blockDim.x + threadIdx.x;
  if (i >= n) return;
  const float p0  = sigmoidf(a[i]);
  const float p1  = sigmoidf(b[i]) * (1.f - p0);
  const float pm1 = 1.f - p0 - p1;
  const float m   = p1 - pm1;
  wm[i] = m;
  wv[i] = p1 + pm1 - m * m;
}

// One workgroup per (batch, class): reduce over K=8192.
__global__ void fc_kernel(const float* __restrict__ e, const float* __restrict__ var_s,
                          const float* __restrict__ wm, const float* __restrict__ wv,
                          float* __restrict__ out) {
  __shared__ float s1[256];
  __shared__ float s2[256];
  const int bo = blockIdx.x;       // b*10 + o
  const int b  = bo / 10;
  const int o  = bo - b * 10;
  const float* eb  = e     + (size_t)b * 8192;
  const float* vb  = var_s + (size_t)b * 8192;
  const float* wmo = wm + (size_t)o * 8192;
  const float* wvo = wv + (size_t)o * 8192;
  float accm = 0.f, accv = 0.f;
  for (int k = threadIdx.x; k < 8192; k += 256) {
    const float ev = eb[k];
    const float vs = vb[k];
    const float m1 = wmo[k];
    const float v1 = wvo[k];
    const float w2 = v1 + m1 * m1;
    const float e2 = ev * ev;
    accm += ev * m1;
    accv += (vs + e2) * w2 - e2 * (m1 * m1);
  }
  s1[threadIdx.x] = accm;
  s2[threadIdx.x] = accv;
  __syncthreads();
  for (int s = 128; s > 0; s >>= 1) {
    if (threadIdx.x < (unsigned)s) {
      s1[threadIdx.x] += s1[threadIdx.x + s];
      s2[threadIdx.x] += s2[threadIdx.x + s];
    }
    __syncthreads();
  }
  if (threadIdx.x == 0) {
    const float m_fc = s1[0];
    const float v_fc = s2[0];
    const float eps  = gaussf(0xabcdef01u, (uint32_t)bo);
    out[bo] = m_fc + sqrtf(fmaxf(v_fc, 1e-10f)) * eps;
  }
}

// ---------------------------------------------------------------------------
// Host orchestration
// ---------------------------------------------------------------------------
extern "C" void kernel_launch(void* const* d_in, const int* in_sizes, int n_in,
                              void* d_out, int out_size, void* d_ws, size_t ws_size,
                              hipStream_t stream) {
  (void)in_sizes; (void)n_in; (void)out_size; (void)ws_size;

  const float* x = (const float*)d_in[0];
  const float* A[6];
  const float* Bt[6];
  const float* G[6];
  for (int i = 0; i < 6; ++i) {
    A[i]  = (const float*)d_in[1 + 3 * i];
    Bt[i] = (const float*)d_in[2 + 3 * i];
    G[i]  = (const float*)d_in[3 + 3 * i];
  }
  const float* bn_gamma = (const float*)d_in[19];
  const float* bn_beta  = (const float*)d_in[20];
  const float* a_fc     = (const float*)d_in[21];
  const float* b_fc     = (const float*)d_in[22];
  float* out = (float*)d_out;

  // Workspace carve-up (~61 MB total; activation ping-pong in bf16).
  char* ws = (char*)d_ws;
  size_t off = 0;
  auto take = [&](size_t bytes) -> char* {
    char* p = ws + off;
    off = (off + bytes + 255) & ~((size_t)255);
    return p;
  };
  uint16_t* bufA = (uint16_t*)take(33554432);  // 128*128*32*32 bf16 (max odd-layer act)
  uint16_t* bufB = (uint16_t*)take(8388608);   // even-layer acts
  float*    m6   = (float*)take(4194304);      // 128*512*4*4 f32 -> later e
  float*    v6   = (float*)take(4194304);      // 128*512*4*4 f32 -> later var_s
  uint32_t* wpM  = (uint32_t*)take(4718592);   // packed A fragments, mean (max layer 6)
  uint32_t* wpV  = (uint32_t*)take(4718592);   // packed A fragments, var
  float*    fwm  = (float*)take(327680);       // FC wm (10x8192)
  float*    fwv  = (float*)take(327680);       // FC wv
  float*    stats= (float*)take(4096);         // mu[512], sd2[512]

  struct LP { int Cin, H, W, Cout, stride; };
  const LP lp[6] = {{3, 32, 32, 128, 1}, {128, 32, 32, 128, 2},
                    {128, 16, 16, 256, 1}, {256, 16, 16, 256, 2},
                    {256, 8, 8, 512, 1},   {512, 8, 8, 512, 2}};
  const int N = 128;

  const void* ins[6]  = {(const void*)x, bufA, bufB, bufA, bufB, bufA};
  uint16_t*   outs[6] = {bufA, bufB, bufA, bufB, bufA, nullptr};

  for (int li = 0; li < 6; ++li) {
    const LP& p = lp[li];
    const int outH = p.H / p.stride;
    const int outW = p.W / p.stride;
    const int Kpad = (p.Cin * 9 + 31) & ~31;
    const int dwords = p.Cout * Kpad / 2;
    pack_weights_kernel<<<dim3((dwords + 255) / 256), dim3(256), 0, stream>>>(
        A[li], Bt[li], wpM, wpV, p.Cout, p.Cin, Kpad);

    const int nCols = N * outH * outW;
    dim3 grid(nCols / 16, p.Cout / 128);   // 4 waves x 2 M-tiles = 128 rows/block
    dim3 blk(128);
    const uint32_t seed = 0x1000u + (uint32_t)li * 0x9e37u;
    if (li == 0) {
      lrconv_wmma_kernel<float, false><<<grid, blk, 0, stream>>>(
          (const float*)ins[li], wpM, wpV, G[li], outs[li], nullptr, nullptr,
          p.Cin, p.H, p.W, p.Cout, outH, outW, p.stride, seed);
    } else if (li == 5) {
      lrconv_wmma_kernel<uint16_t, true><<<grid, blk, 0, stream>>>(
          (const uint16_t*)ins[li], wpM, wpV, G[li], nullptr, m6, v6,
          p.Cin, p.H, p.W, p.Cout, outH, outW, p.stride, seed);
    } else {
      lrconv_wmma_kernel<uint16_t, false><<<grid, blk, 0, stream>>>(
          (const uint16_t*)ins[li], wpM, wpV, G[li], outs[li], nullptr, nullptr,
          p.Cin, p.H, p.W, p.Cout, outH, outW, p.stride, seed);
    }
  }

  const int nFinal = 128 * 512 * 16;  // 1,048,576
  sign_e_kernel<<<dim3(nFinal / 256), dim3(256), 0, stream>>>(m6, v6, nFinal);
  bn_stats_kernel<<<dim3(512), dim3(256), 0, stream>>>(m6, stats);
  bn_apply_kernel<<<dim3(nFinal / 256), dim3(256), 0, stream>>>(
      m6, v6, stats, bn_gamma, bn_beta, nFinal);
  fc_weights_kernel<<<dim3((81920 + 255) / 256), dim3(256), 0, stream>>>(
      a_fc, b_fc, fwm, fwv, 81920);
  fc_kernel<<<dim3(1280), dim3(256), 0, stream>>>(m6, v6, fwm, fwv, out);
}
